// SlidingWindowTransformerProcessor_180388626495
// MI455X (gfx1250) — matrix operand order repaired and checked by reading, not compiled, not run on hardware
//
#include <hip/hip_runtime.h>
#include <hip/hip_bf16.h>

// ---------------------------------------------------------------------------
// Types / helpers
// ---------------------------------------------------------------------------
typedef __bf16 bf16t;
typedef __attribute__((ext_vector_type(16))) __bf16 v16bf;
typedef __attribute__((ext_vector_type(8)))  __bf16 v8bf;
typedef __attribute__((ext_vector_type(8)))  float  v8f;
typedef int v4i_t __attribute__((vector_size(16)));   // b128 payload type

#define DEV static __device__ __forceinline__
#define AS1 __attribute__((address_space(1)))
#define AS3 __attribute__((address_space(3)))

#if defined(__has_builtin)
#if __has_builtin(__builtin_amdgcn_global_load_async_to_lds_b128)
#define HAVE_ASYNC_LDS 1
#endif
#endif
#ifndef HAVE_ASYNC_LDS
#define HAVE_ASYNC_LDS 0
#endif

DEV float bf2f(bf16t b) {
    unsigned short s = __builtin_bit_cast(unsigned short, b);
    unsigned u = ((unsigned)s) << 16;
    return __builtin_bit_cast(float, u);
}
DEV bf16t f2bf(float f) {
    unsigned u = __builtin_bit_cast(unsigned, f);
    unsigned r = (u + 0x7FFFu + ((u >> 16) & 1u)) >> 16;   // round-to-nearest-even
    return __builtin_bit_cast(bf16t, (unsigned short)r);
}
DEV float gelu_exact(float x) { return 0.5f * x * (1.0f + erff(x * 0.70710678118654752f)); }

// async global->LDS staging of one 32-element (64B) bf16 row chunk
DEV void stage_row(const bf16t* __restrict__ g, bf16t* l)
{
#if HAVE_ASYNC_LDS
#pragma unroll
    for (int c = 0; c < 4; c++)
        __builtin_amdgcn_global_load_async_to_lds_b128(
            (AS1 v4i_t*)(g + c * 8), (AS3 v4i_t*)(l + c * 8), 0, 0);
#else
#pragma unroll
    for (int c = 0; c < 4; c++)
        *(v8bf*)(l + c * 8) = *(const v8bf*)(g + c * 8);
#endif
}

DEV void wait_async0()
{
#if HAVE_ASYNC_LDS
#if __has_builtin(__builtin_amdgcn_s_wait_asynccnt)
    __builtin_amdgcn_s_wait_asynccnt(0);
#else
    asm volatile("s_wait_asynccnt 0x0" ::: "memory");
#endif
#endif
}

// Problem constants (match reference)
#define CN   16384
#define CH   512
#define CT   4
#define CE   262144
#define CNH  8
#define CD   2
#define CFF  2048
#define CNT  (CN * CT)          // 65536 flattened rows
#define LPAD 40                 // LDS row pitch (elements): 80B -> conflict-free

// ---------------------------------------------------------------------------
// WMMA bf16 GEMM:  C[M,Nn] = A[M,K] * W[Nn,K]^T + bias
// block = 128 threads (4 waves, 2x2), block tile 128x128, wave tile 64x64
// (4x4 accumulators of v_wmma_f32_16x16x32_bf16).
// A and B K-slabs are double-buffered in LDS, filled with async global->LDS
// copies (ASYNCcnt) so the copy of slab k+1 overlaps the WMMAs of slab k.
// ---------------------------------------------------------------------------
enum { EPI_F32 = 0, EPI_BF16 = 1, EPI_GELU_BF16 = 2 };

template <int EPI>
__global__ __launch_bounds__(128)
void gemm_bf16_wmma(const bf16t* __restrict__ A, const bf16t* __restrict__ W,
                    const float* __restrict__ bias,
                    float* __restrict__ Cf, bf16t* __restrict__ Cb,
                    int M, int Nn, int K)
{
    __shared__ bf16t lA[2][128 * LPAD];
    __shared__ bf16t lB[2][128 * LPAD];

    const int tid  = threadIdx.x;
    const int lane = tid & 31;
    const int wave = tid >> 5;
    const int wm = wave >> 1, wn = wave & 1;
    const int m0 = blockIdx.y * 128 + wm * 64;
    const int n0 = blockIdx.x * 128 + wn * 64;
    const int lr = lane & 15;     // row-within-tile this lane reads
    const int lh = lane >> 4;     // K-half selector (CDNA5 WMMA lane layout)
    const size_t sK = (size_t)K;

    // this thread's staging rows (one A row + one B row of the block tile)
    const bf16t* gA = A + (size_t)(blockIdx.y * 128 + tid) * sK;
    const bf16t* gB = W + (size_t)(blockIdx.x * 128 + tid) * sK;

    v8f acc[4][4];
    v8f z = {};
#pragma unroll
    for (int i = 0; i < 4; i++)
#pragma unroll
        for (int j = 0; j < 4; j++) acc[i][j] = z;

    const int KT = K >> 5;
    // prologue: stage slab 0 into buffer 0
    stage_row(gA, &lA[0][tid * LPAD]);
    stage_row(gB, &lB[0][tid * LPAD]);

    for (int kt = 0; kt < KT; kt++) {
        const int cur = kt & 1;
        wait_async0();        // this wave's async copies into lX[cur] done
        __syncthreads();      // all waves' copies published / prior reads done

        if (kt + 1 < KT) {    // overlap: issue next slab into the other buffer
            const int k1 = (kt + 1) << 5;
            stage_row(gA + k1, &lA[cur ^ 1][tid * LPAD]);
            stage_row(gB + k1, &lB[cur ^ 1][tid * LPAD]);
        }

        // A fragments, 16x32 bf16: lanes 0-15 hold K {0..7,16..23}, lanes
        // 16-31 hold K {8..15,24..31} -> two 16B LDS loads per lane.
        v16bf af[4], bfm[4];
#pragma unroll
        for (int i = 0; i < 4; i++) {
            const bf16t* ar = &lA[cur][(wm * 64 + i * 16 + lr) * LPAD];
            union { v16bf v; v8bf h[2]; } u;
            u.h[0] = *(const v8bf*)(ar + lh * 8);
            u.h[1] = *(const v8bf*)(ar + lh * 8 + 16);
            af[i] = u.v;
        }
        // B fragments, 32x16 = W^T tile: lane (lh,lr) holds W row
        // (n0+j*16+lr), K {lh*8..lh*8+7, 16+lh*8..23+lh*8} -> two 16B LDS loads.
#pragma unroll
        for (int j = 0; j < 4; j++) {
            const bf16t* br = &lB[cur][(wn * 64 + j * 16 + lr) * LPAD];
            union { v16bf v; v8bf h[2]; } u;
            u.h[0] = *(const v8bf*)(br + lh * 8);
            u.h[1] = *(const v8bf*)(br + lh * 8 + 16);
            bfm[j] = u.v;
        }

#pragma unroll
        for (int i = 0; i < 4; i++)
#pragma unroll
            for (int j = 0; j < 4; j++)
                acc[i][j] = __builtin_amdgcn_wmma_f32_16x16x32_bf16(
                    false, af[i], false, bfm[j], (short)0, acc[i][j], false, false);
    }

    // Epilogue. C/D layout: VGPR r -> M = r (lanes 0-15) / 8+r (lanes 16-31),
    // N = lane&15.
#pragma unroll
    for (int i = 0; i < 4; i++) {
        const int rb = m0 + i * 16 + lh * 8;
#pragma unroll
        for (int j = 0; j < 4; j++) {
            const int col = n0 + j * 16 + lr;
            const float bv = bias[col];
#pragma unroll
            for (int r = 0; r < 8; r++) {
                float v = acc[i][j][r] + bv;
                size_t idx = (size_t)(rb + r) * (size_t)Nn + col;
                if constexpr (EPI == EPI_F32)       Cf[idx] = v;
                else if constexpr (EPI == EPI_BF16) Cb[idx] = f2bf(v);
                else                                Cb[idx] = f2bf(gelu_exact(v));
            }
        }
    }
}

// ---------------------------------------------------------------------------
// Small supporting kernels
// ---------------------------------------------------------------------------
__global__ void cvt_f32_bf16(const float* __restrict__ src, bf16t* __restrict__ dst, size_t cnt)
{
    size_t i = (size_t)blockIdx.x * 256 + threadIdx.x;
    if (i < cnt) dst[i] = f2bf(src[i]);
}

__global__ void build_xseq(const float* __restrict__ xmesh, const float* __restrict__ hist,
                           const float* __restrict__ pe,
                           float* __restrict__ xq, bf16t* __restrict__ xb)
{
    size_t i = (size_t)blockIdx.x * 256 + threadIdx.x;   // over NT*H
    if (i >= (size_t)CNT * CH) return;
    int h = (int)(i & (CH - 1));
    int t = (int)((i >> 9) & (CT - 1));
    size_t n = i >> 11;
    float v = (t < CT - 1 ? hist[(n * (CT - 1) + t) * CH + h] : xmesh[n * CH + h]) + pe[t * CH + h];
    xq[i] = v;
    xb[i] = f2bf(v);
}

__global__ void edge_weights(const float* __restrict__ ea, const long long* __restrict__ dst,
                             float* __restrict__ w, float* __restrict__ denom)
{
    int e = blockIdx.x * 256 + threadIdx.x;
    if (e >= CE) return;
    float a = ea[e * 3 + 0], b = ea[e * 3 + 1], c = ea[e * 3 + 2];
    float fn = sqrtf(a * a + b * b + c * c + 1e-12f);
    float wv = expf(-4.0f * fn);
    w[e] = wv;
    atomicAdd(&denom[(int)dst[e]], wv);
}

// one block per edge: agg[dst, 0:2048] += w[e] * x[src, 0:2048]  (all T at once)
__global__ __launch_bounds__(256)
void scatter_agg(const float* __restrict__ xq, const long long* __restrict__ src,
                 const long long* __restrict__ dst, const float* __restrict__ w,
                 float* __restrict__ agg)
{
    int e = blockIdx.x;
    long long s = src[e], d = dst[e];
    float wv = w[e];
    const float* xr = xq + (size_t)s * (CT * CH);
    float* ar = agg + (size_t)d * (CT * CH);
    for (int i = threadIdx.x; i < CT * CH; i += 256)
        atomicAdd(&ar[i], wv * xr[i]);
}

__global__ void agg_finish(const float* __restrict__ agg, const float* __restrict__ denom,
                           bf16t* __restrict__ outb)
{
    size_t i = (size_t)blockIdx.x * 256 + threadIdx.x;   // over NT*H
    if (i >= (size_t)CNT * CH) return;
    int node = (int)(i >> 11);                            // / (T*H)
    float dn = fmaxf(denom[node], 1e-6f);
    outb[i] = f2bf(agg[i] / dn);
}

// one thread per (node, head, tq): causal attention over T=4, d=64
__global__ __launch_bounds__(128)
void attn_kernel(const bf16t* __restrict__ qkv, bf16t* __restrict__ outp)
{
    int gid = blockIdx.x * 128 + threadIdx.x;
    if (gid >= CN * CNH * CT) return;
    int tq = gid & 3;
    int hd = (gid >> 2) & 7;
    size_t n = (size_t)(gid >> 5);
    const bf16t* base = qkv + n * CT * (3 * CH);

    float q[64];
#pragma unroll
    for (int dd = 0; dd < 64; dd++) q[dd] = bf2f(base[(size_t)tq * (3 * CH) + hd * 64 + dd]);

    float sc[4];
#pragma unroll
    for (int tk = 0; tk < 4; tk++) {
        const bf16t* kr = base + (size_t)tk * (3 * CH) + CH + hd * 64;
        float sdot = 0.0f;
#pragma unroll
        for (int dd = 0; dd < 64; dd++) sdot += q[dd] * bf2f(kr[dd]);
        sc[tk] = (tk <= tq) ? sdot * 0.125f : -3.0e38f;   // causal mask
    }
    float mx = fmaxf(fmaxf(sc[0], sc[1]), fmaxf(sc[2], sc[3]));
    float se = 0.0f;
#pragma unroll
    for (int tk = 0; tk < 4; tk++) { sc[tk] = __expf(sc[tk] - mx); se += sc[tk]; }
    float inv = 1.0f / se;

    float o[64];
#pragma unroll
    for (int dd = 0; dd < 64; dd++) o[dd] = 0.0f;
#pragma unroll
    for (int tk = 0; tk < 4; tk++) {
        const bf16t* vr = base + (size_t)tk * (3 * CH) + 2 * CH + hd * 64;
        float p = sc[tk] * inv;
#pragma unroll
        for (int dd = 0; dd < 64; dd++) o[dd] += p * bf2f(vr[dd]);
    }
    bf16t* orow = outp + (n * CT + tq) * CH + hd * 64;
#pragma unroll
    for (int dd = 0; dd < 64; dd++) orow[dd] = f2bf(o[dd]);
}

// one wave (32 lanes) per row of 512: x = LN(xin + delta)*g + b -> f32 + bf16
__global__ __launch_bounds__(256)
void ln_kernel(const float* __restrict__ xin, const float* __restrict__ delta,
               const float* __restrict__ g, const float* __restrict__ b,
               float* __restrict__ xoutF, bf16t* __restrict__ xoutB)
{
    int row = blockIdx.x * 8 + (threadIdx.x >> 5);
    int lane = threadIdx.x & 31;
    const float* xr = xin + (size_t)row * CH;
    const float* dr = delta + (size_t)row * CH;

    float v[16];
    float s = 0.0f, ss = 0.0f;
#pragma unroll
    for (int i = 0; i < 16; i++) {
        float t = xr[lane * 16 + i] + dr[lane * 16 + i];
        v[i] = t; s += t; ss += t * t;
    }
#pragma unroll
    for (int off = 16; off > 0; off >>= 1) {
        s  += __shfl_xor(s, off, 32);
        ss += __shfl_xor(ss, off, 32);
    }
    float mean = s * (1.0f / CH);
    float var  = ss * (1.0f / CH) - mean * mean;
    float inv  = rsqrtf(var + 1e-5f);
#pragma unroll
    for (int i = 0; i < 16; i++) {
        int c = lane * 16 + i;
        float o = (v[i] - mean) * inv * g[c] + b[c];
        xoutF[(size_t)row * CH + c] = o;
        xoutB[(size_t)row * CH + c] = f2bf(o);
    }
}

__global__ void final_slice(const float* __restrict__ xq, float* __restrict__ out)
{
    size_t i = (size_t)blockIdx.x * 256 + threadIdx.x;   // over N*H
    if (i >= (size_t)CN * CH) return;
    size_t n = i >> 9;
    int h = (int)(i & (CH - 1));
    out[i] = xq[(n * CT + (CT - 1)) * CH + h];
}

// ---------------------------------------------------------------------------
// Host launch
// ---------------------------------------------------------------------------
extern "C" void kernel_launch(void* const* d_in, const int* /*in_sizes*/, int /*n_in*/,
                              void* d_out, int /*out_size*/, void* d_ws, size_t /*ws_size*/,
                              hipStream_t stream)
{
    const float* x_mesh  = (const float*)d_in[0];
    const float* hist    = (const float*)d_in[1];
    const float* eattr   = (const float*)d_in[2];
    const float* pe      = (const float*)d_in[3];
    const float* qkv_w   = (const float*)d_in[4];
    const float* qkv_b   = (const float*)d_in[5];
    const float* out_w   = (const float*)d_in[6];
    const float* out_b   = (const float*)d_in[7];
    const float* ff_w1   = (const float*)d_in[8];
    const float* ff_b1   = (const float*)d_in[9];
    const float* ff_w2   = (const float*)d_in[10];
    const float* ff_b2   = (const float*)d_in[11];
    const float* ln1_g   = (const float*)d_in[12];
    const float* ln1_b   = (const float*)d_in[13];
    const float* ln2_g   = (const float*)d_in[14];
    const float* ln2_b   = (const float*)d_in[15];
    const float* mix_w1  = (const float*)d_in[16];
    const float* mix_b1  = (const float*)d_in[17];
    const float* mix_w2  = (const float*)d_in[18];
    const float* mix_b2  = (const float*)d_in[19];
    const float* mixln_g = (const float*)d_in[20];
    const float* mixln_b = (const float*)d_in[21];
    const long long* eidx = (const long long*)d_in[22];
    const long long* esrc = eidx;
    const long long* edst = eidx + CE;
    float* outp = (float*)d_out;

    // ---- workspace carve (256B aligned) ----
    char* base = (char*)d_ws;
    size_t off = 0;
    auto carve = [&](size_t bytes) -> void* {
        void* p = base + off;
        off = (off + bytes + 255) & ~(size_t)255;
        return p;
    };
    float* xq    = (float*)carve((size_t)CNT * CH * 4);          // residual stream f32
    bf16t* xb    = (bf16t*)carve((size_t)CNT * CH * 2);          // GEMM input bf16
    bf16t* qkvb  = (bf16t*)carve((size_t)CNT * 3 * CH * 2);      // qkv / mix hidden
    bf16t* attno = (bf16t*)carve((size_t)CNT * CH * 2);          // attn out / agg bf16
    float* t0    = (float*)carve((size_t)CNT * CH * 4);          // GEMM f32 out / agg
    bf16t* hbuf  = (bf16t*)carve((size_t)CNT * CFF * 2);         // FF hidden
    bf16t* wqkv  = (bf16t*)carve((size_t)CD * 3 * CH * CH * 2);
    bf16t* wout  = (bf16t*)carve((size_t)CD * CH * CH * 2);
    bf16t* wff1  = (bf16t*)carve((size_t)CD * CFF * CH * 2);
    bf16t* wff2  = (bf16t*)carve((size_t)CD * CH * CFF * 2);
    bf16t* wmx1  = (bf16t*)carve((size_t)CH * CH * 2);
    bf16t* wmx2  = (bf16t*)carve((size_t)CH * CH * 2);
    float* ew    = (float*)carve((size_t)CE * 4);
    float* denom = (float*)carve((size_t)CN * 4);

    auto cvt = [&](const float* s, bf16t* d, size_t cnt) {
        cvt_f32_bf16<<<(unsigned)((cnt + 255) / 256), 256, 0, stream>>>(s, d, cnt);
    };

    // ---- weights -> bf16 ----
    cvt(qkv_w, wqkv, (size_t)CD * 3 * CH * CH);
    cvt(out_w, wout, (size_t)CD * CH * CH);
    cvt(ff_w1, wff1, (size_t)CD * CFF * CH);
    cvt(ff_w2, wff2, (size_t)CD * CH * CFF);
    cvt(mix_w1, wmx1, (size_t)CH * CH);
    cvt(mix_w2, wmx2, (size_t)CH * CH);

    // ---- edge weights + denom ----
    (void)hipMemsetAsync(denom, 0, (size_t)CN * 4, stream);
    edge_weights<<<(CE + 255) / 256, 256, 0, stream>>>(eattr, edst, ew, denom);

    // ---- x_seq = concat(hist, x) + pe ----
    build_xseq<<<(unsigned)(((size_t)CNT * CH + 255) / 256), 256, 0, stream>>>(x_mesh, hist, pe, xq, xb);

    const unsigned gM = CNT / 128;                    // 512 M-tiles
    const unsigned elemBlocks = (unsigned)(((size_t)CNT * CH + 255) / 256);

    for (int d = 0; d < CD; d++) {
        // QKV projection: [NT,512] x [1536,512]^T -> bf16 [NT,1536]
        gemm_bf16_wmma<EPI_BF16><<<dim3(3 * CH / 128, gM), 128, 0, stream>>>(
            xb, wqkv + (size_t)d * 3 * CH * CH, qkv_b + (size_t)d * 3 * CH,
            nullptr, qkvb, CNT, 3 * CH, CH);

        // causal MHA over T=4
        attn_kernel<<<(CN * CNH * CT) / 128, 128, 0, stream>>>(qkvb, attno);

        // out projection -> f32 t0
        gemm_bf16_wmma<EPI_F32><<<dim3(CH / 128, gM), 128, 0, stream>>>(
            attno, wout + (size_t)d * CH * CH, out_b + (size_t)d * CH,
            t0, nullptr, CNT, CH, CH);

        // y = LN1(x + attn)
        ln_kernel<<<CNT / 8, 256, 0, stream>>>(xq, t0, ln1_g + (size_t)d * CH, ln1_b + (size_t)d * CH, xq, xb);

        // FF1 + GELU -> bf16 hidden
        gemm_bf16_wmma<EPI_GELU_BF16><<<dim3(CFF / 128, gM), 128, 0, stream>>>(
            xb, wff1 + (size_t)d * CFF * CH, ff_b1 + (size_t)d * CFF,
            nullptr, hbuf, CNT, CFF, CH);

        // FF2 -> f32 t0
        gemm_bf16_wmma<EPI_F32><<<dim3(CH / 128, gM), 128, 0, stream>>>(
            hbuf, wff2 + (size_t)d * CH * CFF, ff_b2 + (size_t)d * CH,
            t0, nullptr, CNT, CH, CFF);

        // x = LN2(y + ff)
        ln_kernel<<<CNT / 8, 256, 0, stream>>>(xq, t0, ln2_g + (size_t)d * CH, ln2_b + (size_t)d * CH, xq, xb);

        // ---- spatial mix (all 4 timesteps batched; shared weights) ----
        (void)hipMemsetAsync(t0, 0, (size_t)CNT * CH * 4, stream);           // t0 = agg
        scatter_agg<<<CE, 256, 0, stream>>>(xq, esrc, edst, ew, t0);
        agg_finish<<<elemBlocks, 256, 0, stream>>>(t0, denom, attno);        // attno = agg/denom (bf16)

        gemm_bf16_wmma<EPI_GELU_BF16><<<dim3(CH / 128, gM), 128, 0, stream>>>(
            attno, wmx1, mix_b1, nullptr, qkvb, CNT, CH, CH);                // qkvb = gelu(agg W1^T + b1)
        gemm_bf16_wmma<EPI_F32><<<dim3(CH / 128, gM), 128, 0, stream>>>(
            qkvb, wmx2, mix_b2, t0, nullptr, CNT, CH, CH);                   // t0 = msg

        // x = mixLN(x + msg)
        ln_kernel<<<CNT / 8, 256, 0, stream>>>(xq, t0, mixln_g, mixln_b, xq, xb);
    }

    // output = x_seq[:, -1, :]
    final_slice<<<(unsigned)(((size_t)CN * CH + 255) / 256), 256, 0, stream>>>(xq, outp);
}
